// CompositionalKoopmanOperators_66417374265730
// MI455X (gfx1250) — matrix operand order repaired and checked by reading, not compiled, not run on hardware
//
#include <hip/hip_runtime.h>

typedef __attribute__((ext_vector_type(16))) _Float16 v16h;
typedef __attribute__((ext_vector_type(8)))  float    v8f;

// ---------------- optional CDNA5 async global->LDS path ---------------------
#if __has_builtin(__builtin_amdgcn_global_load_async_to_lds_b128)
#define HAVE_ASYNC_LDS 1
typedef __attribute__((__vector_size__(4 * sizeof(int)))) int v4i_t;
typedef __attribute__((address_space(1))) v4i_t glb_v4i;
typedef __attribute__((address_space(3))) v4i_t lds_v4i;
__device__ __forceinline__ void async_copy16(const void* g, void* l) {
  __builtin_amdgcn_global_load_async_to_lds_b128(
      (glb_v4i*)(unsigned long long)g,
      (lds_v4i*)(unsigned)(unsigned long long)l, 0, 0);
}
__device__ __forceinline__ void async_wait0() {
#if __has_builtin(__builtin_amdgcn_s_wait_asynccnt)
  __builtin_amdgcn_s_wait_asynccnt(0);
#else
  asm volatile("s_wait_asynccnt 0x0" ::: "memory");
#endif
}
#endif

// ---------------------------------------------------------------------------
// D = A(16xK, LDS, f16) * W^T-tile (Kx16 from row-major [out][in] f16 weights)
// accumulated into v8f per CDNA5 16x16x32 f16 WMMA layouts.
// A frag: lane row = lane&15, k = kk + (v>>2)*16 + (lane>>4)*8 + (v&3)*2
// B frag: lane col = lane&15, k = kk + (v>>2)*16 + (lane>>4)*8 + (v&3)*2
// ---------------------------------------------------------------------------
template <int K>
__device__ __forceinline__ v8f wmma_tile(const _Float16* A, int lda,
                                         const _Float16* W, int ldw,
                                         int m0, int n0, v8f acc) {
  const int lane = threadIdx.x & 31;
  const int r  = lane & 15;
  const int hl = lane >> 4;
  const _Float16* arow = A + (m0 + r) * lda;
  const _Float16* wrow = W + (n0 + r) * ldw;
#pragma unroll
  for (int kk = 0; kk < K; kk += 32) {
    union { v16h h; unsigned u[8]; } a, b;
#pragma unroll
    for (int v = 0; v < 8; ++v) {
      a.u[v] = *(const unsigned*)(arow + kk + ((v >> 2) << 4) + (hl << 3) + ((v & 3) << 1));
      b.u[v] = *(const unsigned*)(wrow + kk + ((v >> 2) << 4) + (hl << 3) + ((v & 3) << 1));
    }
    acc = __builtin_amdgcn_wmma_f32_16x16x32_f16(
        false, a.h, false, b.h, (short)0, acc, false, false);
  }
  return acc;
}

// -------------------------- weight fp32 -> f16 (with K padding) ------------
__global__ void cvt_pad_kernel(const float* __restrict__ src, _Float16* __restrict__ dst,
                               int rows, int sc, int dc) {
  int i = blockIdx.x * 256 + threadIdx.x;
  if (i >= rows * dc) return;
  int r = i / dc, c = i - r * dc;
  dst[i] = (c < sc) ? (_Float16)src[r * sc + c] : (_Float16)0.f;
}

// -------------------------- object encoder ---------------------------------
// rows = B*N = 2048 ; [attr(4)|state(8)|pad] -> 128 -> 128, relu both
__global__ void obj_encode_kernel(const float* __restrict__ attrs,
                                  const float* __restrict__ states,
                                  const _Float16* __restrict__ w0, const float* __restrict__ b0,
                                  const _Float16* __restrict__ w1, const float* __restrict__ b1,
                                  _Float16* __restrict__ objh) {
  __shared__ __align__(16) _Float16 A1[64 * 40];
  __shared__ __align__(16) _Float16 A2[64 * 136];
  const int m0 = blockIdx.x * 64;
  const int t = threadIdx.x;
  for (int idx = t; idx < 64 * 32; idx += 256) {
    int r = idx >> 5, c = idx & 31;
    int row = m0 + r;
    float v = 0.f;
    if (c < 4)       v = attrs[row * 4 + c];
    else if (c < 12) v = states[row * 8 + (c - 4)];
    A1[r * 40 + c] = (_Float16)v;
  }
  __syncthreads();
  const int w = t >> 5, lane = t & 31, rr = lane & 15, hl = lane >> 4;
  const int n = w * 16 + rr;
  const float bias0 = b0[n];
#pragma unroll
  for (int mt = 0; mt < 4; ++mt) {
    v8f acc = {};
    acc = wmma_tile<32>(A1, 40, w0, 32, mt * 16, w * 16, acc);
#pragma unroll
    for (int v = 0; v < 8; ++v) {
      int m = mt * 16 + v + 8 * hl;
      float x = acc[v] + bias0;
      A2[m * 136 + n] = (_Float16)(x > 0.f ? x : 0.f);
    }
  }
  __syncthreads();
  const float bias1 = b1[n];
#pragma unroll
  for (int mt = 0; mt < 4; ++mt) {
    v8f acc = {};
    acc = wmma_tile<128>(A2, 136, w1, 128, mt * 16, w * 16, acc);
#pragma unroll
    for (int v = 0; v < 8; ++v) {
      int m = mt * 16 + v + 8 * hl;
      float x = acc[v] + bias1;
      objh[(m0 + m) * 128 + n] = (_Float16)(x > 0.f ? x : 0.f);
    }
  }
}

// -------------------------- relation encoder --------------------------------
// one block per (b,i); rows j=0..63 ; 20 feats -> 128 -> 128 relu
__global__ void rel_encode_kernel(const float* __restrict__ attrs,
                                  const float* __restrict__ states,
                                  const float* __restrict__ rel_attrs,
                                  const _Float16* __restrict__ w0, const float* __restrict__ b0,
                                  const _Float16* __restrict__ w1, const float* __restrict__ b1,
                                  _Float16* __restrict__ relh) {
  __shared__ __align__(16) _Float16 A1[64 * 40];
  __shared__ __align__(16) _Float16 A2[64 * 136];
  const int blk = blockIdx.x;      // b*64 + i
  const int b = blk >> 6;
  const int t = threadIdx.x;
  // warm weight lines (L2-resident across 2048 blocks)
  for (int p = t; p < (128 * 128 * 2) / 128; p += 256)
    __builtin_prefetch(((const char*)w1) + p * 128, 0, 1);
  for (int idx = t; idx < 64 * 32; idx += 256) {
    int j = idx >> 5, c = idx & 31;
    float v = 0.f;
    if (c < 4)       v = rel_attrs[(blk * 64 + j) * 4 + c];
    else if (c < 12) v = states[blk * 8 + (c - 4)] - states[(b * 64 + j) * 8 + (c - 4)];
    else if (c < 16) v = attrs[blk * 4 + (c - 12)];
    else if (c < 20) v = attrs[(b * 64 + j) * 4 + (c - 16)];
    A1[j * 40 + c] = (_Float16)v;
  }
  __syncthreads();
  const int w = t >> 5, lane = t & 31, rr = lane & 15, hl = lane >> 4;
  const int n = w * 16 + rr;
  const float bias0 = b0[n];
#pragma unroll
  for (int mt = 0; mt < 4; ++mt) {
    v8f acc = {};
    acc = wmma_tile<32>(A1, 40, w0, 32, mt * 16, w * 16, acc);
#pragma unroll
    for (int v = 0; v < 8; ++v) {
      int m = mt * 16 + v + 8 * hl;
      float x = acc[v] + bias0;
      A2[m * 136 + n] = (_Float16)(x > 0.f ? x : 0.f);
    }
  }
  __syncthreads();
  const float bias1 = b1[n];
#pragma unroll
  for (int mt = 0; mt < 4; ++mt) {
    v8f acc = {};
    acc = wmma_tile<128>(A2, 136, w1, 128, mt * 16, w * 16, acc);
#pragma unroll
    for (int v = 0; v < 8; ++v) {
      int m = mt * 16 + v + 8 * hl;
      float x = acc[v] + bias1;
      relh[(blk * 64 + m) * 128 + n] = (_Float16)(x > 0.f ? x : 0.f);
    }
  }
}

// -------------------------- propagation: edge effect + aggregate ------------
// one block per (b,i); [rel(128)|obj_i(128)|obj_j(128)] x rp_w -> LN -> relu -> sum_j
__global__ void prop_rel_kernel(const _Float16* __restrict__ relh,
                                const _Float16* __restrict__ objh,
                                const _Float16* __restrict__ wrp, const float* __restrict__ brp,
                                const float* __restrict__ lnw, const float* __restrict__ lnb,
                                float* __restrict__ aggf) {
  __shared__ __align__(16) _Float16 Ah[32 * 392];
  __shared__ float Dh[32 * 132];
  __shared__ float colsum[128];
  __shared__ float ps[32 * 8], ps2[32 * 8];
  __shared__ float mu[32], rs[32];
  const int blk = blockIdx.x;      // b*64 + i
  const int b = blk >> 6;
  const int t = threadIdx.x;
  // warm the 96KB f16 weight panel into cache while A-panel streams in
  for (int p = t; p < (128 * 384 * 2) / 128; p += 256)
    __builtin_prefetch(((const char*)wrp) + p * 128, 0, 1);
  if (t < 128) colsum[t] = 0.f;
  const int w = t >> 5, lane = t & 31, rr = lane & 15, hl = lane >> 4;
  const int n = w * 16 + rr;
  const float bias = brp[n];
  __syncthreads();
  for (int ch = 0; ch < 2; ++ch) {
#ifdef HAVE_ASYNC_LDS
    // 32 rows x 768B, 16B granules: three contiguous 256B segments per row
    for (int idx = t; idx < 32 * 48; idx += 256) {
      int r = idx / 48, s16 = idx - r * 48;
      int j = ch * 32 + r;
      int off = s16 * 16;  // byte offset within logical 768B row
      unsigned long long g =
          (off < 256)
              ? (unsigned long long)(relh + (size_t)(blk * 64 + j) * 128) + off
              : (off < 512)
                    ? (unsigned long long)(objh + (size_t)blk * 128) + (off - 256)
                    : (unsigned long long)(objh + (size_t)(b * 64 + j) * 128) + (off - 512);
      async_copy16((const void*)g, (char*)(Ah + r * 392) + off);
    }
    async_wait0();
#else
    for (int idx = t; idx < 32 * 384; idx += 256) {
      int r = idx / 384, k = idx - r * 384;
      int j = ch * 32 + r;
      _Float16 v;
      if (k < 128)      v = relh[(blk * 64 + j) * 128 + k];
      else if (k < 256) v = objh[blk * 128 + (k - 128)];
      else              v = objh[(b * 64 + j) * 128 + (k - 256)];
      Ah[r * 392 + k] = v;
    }
#endif
    __syncthreads();
#pragma unroll
    for (int mt = 0; mt < 2; ++mt) {
      v8f acc = {};
      acc = wmma_tile<384>(Ah, 392, wrp, 384, mt * 16, w * 16, acc);
#pragma unroll
      for (int v = 0; v < 8; ++v) {
        int m = mt * 16 + v + 8 * hl;
        Dh[m * 132 + n] = acc[v] + bias;
      }
    }
    __syncthreads();
    {
      int r = t >> 3, seg = t & 7;
      float s = 0.f, s2 = 0.f;
      for (int c = seg * 16; c < seg * 16 + 16; ++c) {
        float v = Dh[r * 132 + c];
        s += v; s2 += v * v;
      }
      ps[r * 8 + seg] = s; ps2[r * 8 + seg] = s2;
    }
    __syncthreads();
    if (t < 32) {
      float s = 0.f, s2 = 0.f;
      for (int k = 0; k < 8; ++k) { s += ps[t * 8 + k]; s2 += ps2[t * 8 + k]; }
      float mean = s * (1.f / 128.f);
      float var = s2 * (1.f / 128.f) - mean * mean;
      mu[t] = mean;
      rs[t] = rsqrtf(var + 1e-5f);
    }
    __syncthreads();
    for (int idx = t; idx < 32 * 128; idx += 256) {
      int r = idx >> 7, c = idx & 127;
      float v = (Dh[r * 132 + c] - mu[r]) * rs[r] * lnw[c] + lnb[c];
      Dh[r * 132 + c] = v > 0.f ? v : 0.f;
    }
    __syncthreads();
    if (t < 128) {
      float s = 0.f;
      for (int r = 0; r < 32; ++r) s += Dh[r * 132 + t];
      colsum[t] += s;
    }
    __syncthreads();
  }
  if (t < 128) aggf[blk * 128 + t] = colsum[t];
}

// -------------------------- propagation: object update ----------------------
// rows = 2048 in blocks of 32: [obj|agg](256) x pp_w -> LN -> relu -> objh
__global__ void prop_obj_kernel(_Float16* __restrict__ objh,
                                const float* __restrict__ aggf,
                                const _Float16* __restrict__ wpp, const float* __restrict__ bpp,
                                const float* __restrict__ lnw, const float* __restrict__ lnb) {
  __shared__ __align__(16) _Float16 Ah[32 * 264];
  __shared__ float Dh[32 * 132];
  __shared__ float ps[32 * 8], ps2[32 * 8];
  __shared__ float mu[32], rs[32];
  const int m0 = blockIdx.x * 32;
  const int t = threadIdx.x;
  for (int idx = t; idx < 32 * 256; idx += 256) {
    int r = idx >> 8, k = idx & 255;
    _Float16 v;
    if (k < 128) v = objh[(m0 + r) * 128 + k];
    else         v = (_Float16)aggf[(m0 + r) * 128 + (k - 128)];
    Ah[r * 264 + k] = v;
  }
  __syncthreads();
  const int w = t >> 5, lane = t & 31, rr = lane & 15, hl = lane >> 4;
  const int n = w * 16 + rr;
  const float bias = bpp[n];
#pragma unroll
  for (int mt = 0; mt < 2; ++mt) {
    v8f acc = {};
    acc = wmma_tile<256>(Ah, 264, wpp, 256, mt * 16, w * 16, acc);
#pragma unroll
    for (int v = 0; v < 8; ++v) {
      int m = mt * 16 + v + 8 * hl;
      Dh[m * 132 + n] = acc[v] + bias;
    }
  }
  __syncthreads();
  {
    int r = t >> 3, seg = t & 7;
    float s = 0.f, s2 = 0.f;
    for (int c = seg * 16; c < seg * 16 + 16; ++c) {
      float v = Dh[r * 132 + c];
      s += v; s2 += v * v;
    }
    ps[r * 8 + seg] = s; ps2[r * 8 + seg] = s2;
  }
  __syncthreads();
  if (t < 32) {
    float s = 0.f, s2 = 0.f;
    for (int k = 0; k < 8; ++k) { s += ps[t * 8 + k]; s2 += ps2[t * 8 + k]; }
    float mean = s * (1.f / 128.f);
    float var = s2 * (1.f / 128.f) - mean * mean;
    mu[t] = mean;
    rs[t] = rsqrtf(var + 1e-5f);
  }
  __syncthreads();
  for (int idx = t; idx < 32 * 128; idx += 256) {
    int r = idx >> 7, c = idx & 127;
    float v = (Dh[r * 132 + c] - mu[r]) * rs[r] * lnw[c] + lnb[c];
    objh[(m0 + r) * 128 + c] = (_Float16)(v > 0.f ? v : 0.f);
  }
}

// -------------------------- predictor: 128 -> relu -> 32 -> tanh ------------
__global__ void predict_kernel(const _Float16* __restrict__ objh,
                               const _Float16* __restrict__ w0, const float* __restrict__ b0,
                               const _Float16* __restrict__ w1, const float* __restrict__ b1,
                               float* __restrict__ out) {
  __shared__ __align__(16) _Float16 A1[64 * 136];
  __shared__ __align__(16) _Float16 A2[64 * 136];
  const int m0 = blockIdx.x * 64;
  const int t = threadIdx.x;
#ifdef HAVE_ASYNC_LDS
  for (int idx = t; idx < 64 * 16; idx += 256) {   // 64 rows x 256B in 16B granules
    int r = idx >> 4, s16 = idx & 15;
    async_copy16((const char*)(objh + (size_t)(m0 + r) * 128) + s16 * 16,
                 (char*)(A1 + r * 136) + s16 * 16);
  }
  async_wait0();
#else
  for (int idx = t; idx < 64 * 128; idx += 256) {
    int r = idx >> 7, k = idx & 127;
    A1[r * 136 + k] = objh[(m0 + r) * 128 + k];
  }
#endif
  __syncthreads();
  const int w = t >> 5, lane = t & 31, rr = lane & 15, hl = lane >> 4;
  const int n = w * 16 + rr;
  const float bias0 = b0[n];
#pragma unroll
  for (int mt = 0; mt < 4; ++mt) {
    v8f acc = {};
    acc = wmma_tile<128>(A1, 136, w0, 128, mt * 16, w * 16, acc);
#pragma unroll
    for (int v = 0; v < 8; ++v) {
      int m = mt * 16 + v + 8 * hl;
      float x = acc[v] + bias0;
      A2[m * 136 + n] = (_Float16)(x > 0.f ? x : 0.f);
    }
  }
  __syncthreads();
  if (w < 2) {                          // only 32 output channels
    const float bias1 = b1[n];
#pragma unroll
    for (int mt = 0; mt < 4; ++mt) {
      v8f acc = {};
      acc = wmma_tile<128>(A2, 136, w1, 128, mt * 16, w * 16, acc);
#pragma unroll
      for (int v = 0; v < 8; ++v) {
        int m = mt * 16 + v + 8 * hl;
        out[(m0 + m) * 32 + n] = tanhf(acc[v] + bias1);
      }
    }
  }
}

// ---------------------------------------------------------------------------
extern "C" void kernel_launch(void* const* d_in, const int* in_sizes, int n_in,
                              void* d_out, int out_size, void* d_ws, size_t ws_size,
                              hipStream_t stream) {
  (void)in_sizes; (void)n_in; (void)out_size; (void)ws_size;
  const float* attrs     = (const float*)d_in[0];
  const float* states    = (const float*)d_in[1];
  const float* rel_attrs = (const float*)d_in[3];
  const float* oe_w0 = (const float*)d_in[4];  const float* oe_b0 = (const float*)d_in[5];
  const float* oe_w1 = (const float*)d_in[6];  const float* oe_b1 = (const float*)d_in[7];
  const float* re_w0 = (const float*)d_in[8];  const float* re_b0 = (const float*)d_in[9];
  const float* re_w1 = (const float*)d_in[10]; const float* re_b1 = (const float*)d_in[11];
  const float* rp_w  = (const float*)d_in[12]; const float* rp_b  = (const float*)d_in[13];
  const float* rp_lnw = (const float*)d_in[14]; const float* rp_lnb = (const float*)d_in[15];
  const float* pp_w  = (const float*)d_in[16]; const float* pp_b  = (const float*)d_in[17];
  const float* pp_lnw = (const float*)d_in[18]; const float* pp_lnb = (const float*)d_in[19];
  const float* pr_w0 = (const float*)d_in[20]; const float* pr_b0 = (const float*)d_in[21];
  const float* pr_w1 = (const float*)d_in[22]; const float* pr_b1 = (const float*)d_in[23];
  float* out = (float*)d_out;

  char* ws = (char*)d_ws;
  size_t off = 0;
  auto alloc = [&](size_t bytes) -> void* {
    void* p = ws + off;
    off += (bytes + 255) & ~(size_t)255;
    return p;
  };
  _Float16* w_oe0h = (_Float16*)alloc(128 * 32 * 2);
  _Float16* w_oe1h = (_Float16*)alloc(128 * 128 * 2);
  _Float16* w_re0h = (_Float16*)alloc(128 * 32 * 2);
  _Float16* w_re1h = (_Float16*)alloc(128 * 128 * 2);
  _Float16* w_rph  = (_Float16*)alloc(128 * 384 * 2);
  _Float16* w_pph  = (_Float16*)alloc(128 * 256 * 2);
  _Float16* w_pr0h = (_Float16*)alloc(128 * 128 * 2);
  _Float16* w_pr1h = (_Float16*)alloc(32 * 128 * 2);
  _Float16* objh   = (_Float16*)alloc((size_t)2048 * 128 * 2);
  _Float16* relh   = (_Float16*)alloc((size_t)131072 * 128 * 2);
  float*    aggf   = (float*)   alloc((size_t)2048 * 128 * 4);

  auto cvt = [&](const float* s, _Float16* d, int rows, int sc, int dc) {
    int nel = rows * dc;
    cvt_pad_kernel<<<(nel + 255) / 256, 256, 0, stream>>>(s, d, rows, sc, dc);
  };
  cvt(oe_w0, w_oe0h, 128, 12, 32);
  cvt(oe_w1, w_oe1h, 128, 128, 128);
  cvt(re_w0, w_re0h, 128, 20, 32);
  cvt(re_w1, w_re1h, 128, 128, 128);
  cvt(rp_w,  w_rph,  128, 384, 384);
  cvt(pp_w,  w_pph,  128, 256, 256);
  cvt(pr_w0, w_pr0h, 128, 128, 128);
  cvt(pr_w1, w_pr1h, 32, 128, 128);

  obj_encode_kernel<<<32, 256, 0, stream>>>(attrs, states, w_oe0h, oe_b0, w_oe1h, oe_b1, objh);
  rel_encode_kernel<<<2048, 256, 0, stream>>>(attrs, states, rel_attrs,
                                              w_re0h, re_b0, w_re1h, re_b1, relh);
  for (int step = 0; step < 2; ++step) {   // pstep = 2 (reference constant)
    prop_rel_kernel<<<2048, 256, 0, stream>>>(relh, objh, w_rph, rp_b, rp_lnw, rp_lnb, aggf);
    prop_obj_kernel<<<64, 256, 0, stream>>>(objh, aggf, w_pph, pp_b, pp_lnw, pp_lnb);
  }
  predict_kernel<<<32, 256, 0, stream>>>(objh, w_pr0h, pr_b0, w_pr1h, pr_b1, out);
}
